// MambaBlock_83434034692782
// MI455X (gfx1250) — matrix operand order repaired and checked by reading, not compiled, not run on hardware
//
#include <hip/hip_runtime.h>
#include <hip/hip_bf16.h>
#include <math.h>

typedef __attribute__((ext_vector_type(16))) _Float16 v16h;
typedef __attribute__((ext_vector_type(8)))  _Float16 v8h;
typedef __attribute__((ext_vector_type(8)))  float    v8f;
typedef __attribute__((ext_vector_type(4)))  int      v4i;

#define DMODEL 1024
#define DINNER 2048
#define TLEN   2048
#define BATCH  4
#define NROWS  (BATCH*TLEN)   // 8192

// ---- CDNA5 async global->LDS (confirmed present; params are v4i* w/ AS) ----
#if defined(__AMDGCN__) && __has_builtin(__builtin_amdgcn_global_load_async_to_lds_b128)
#define HAVE_ASYNC_LDS 1
#else
#define HAVE_ASYNC_LDS 0
#endif

#if HAVE_ASYNC_LDS
#define AS1 __attribute__((address_space(1)))
#define AS3 __attribute__((address_space(3)))
__device__ __forceinline__ void async_copy16(const _Float16* g, _Float16* l) {
    __builtin_amdgcn_global_load_async_to_lds_b128(
        (AS1 v4i*)(uintptr_t)g,
        (AS3 v4i*)(unsigned)(uintptr_t)l,
        0, 0);
}
#define ASYNC_WAIT(n) __builtin_amdgcn_s_wait_asynccnt(n)
#endif

// ---------------------------------------------------------------- convert
__global__ __launch_bounds__(256)
void cvt_f32_f16(const float* __restrict__ src, _Float16* __restrict__ dst, int n) {
    int i = blockIdx.x * 256 + threadIdx.x;
    if (i < n) dst[i] = (_Float16)src[i];
}

// ---------------------------------------------------------------- GEMM
// C[M,N] = A[M,K] * Bw[N,K]^T   (row-major A, row-major weight Bw)
// f16 inputs, f32 accumulate via v_wmma_f32_16x16x32_f16.
// Workgroup tile 128(M) x 128(N); 8 waves, each wave 64x32 (4x2 WMMA).
// Double-buffered LDS, async global->LDS staging (ASYNCcnt) when available.
// M,K multiples of 128/32 at all call sites; N guarded (clamped B rows,
// guarded stores) for the N=16 x_proj case.
template<bool HALF_OUT>
__global__ __launch_bounds__(256)
void gemm_f16_wmma(const _Float16* __restrict__ A,
                   const _Float16* __restrict__ Bw,
                   void* __restrict__ Cv,
                   int M, int N, int K) {
    __shared__ __align__(32) _Float16 As[2][128 * 32];  // [m][k]
    __shared__ __align__(32) _Float16 Bs[2][128 * 32];  // [n][k]

    const int tid   = threadIdx.x;
    const int lane  = tid & 31;
    const int wave  = tid >> 5;
    const int waveM = wave >> 2;        // 0..1 -> 64-row slab
    const int waveN = wave & 3;         // 0..3 -> 32-col slab
    const int bm    = blockIdx.y * 128;
    const int bn    = blockIdx.x * 128;
    const int hsel  = lane >> 4;        // lane half (wave32)

    v8f acc[4][2] = {};

    // per-thread staging map: 512 x 16B per tile, 2 per thread
    auto stage = [&](int sel, int k0) {
        #pragma unroll
        for (int it = 0; it < 2; ++it) {
            int li  = tid + 256 * it;          // 0..511
            int row = li >> 2;
            int col = (li & 3) << 3;
            const _Float16* ga = A + (size_t)(bm + row) * K + k0 + col;
            int gn = bn + row;
            const _Float16* gb = Bw + (size_t)(gn < N ? gn : 0) * K + k0 + col;
#if HAVE_ASYNC_LDS
            async_copy16(ga, &As[sel][row * 32 + col]);
            async_copy16(gb, &Bs[sel][row * 32 + col]);
#else
            *(v8h*)(&As[sel][row * 32 + col]) = *(const v8h*)ga;
            *(v8h*)(&Bs[sel][row * 32 + col]) = *(const v8h*)gb;
#endif
        }
    };

    int sel = 0;
    stage(0, 0);
    for (int k0 = 0; k0 < K; k0 += 32) {
        const bool more = (k0 + 32) < K;
        if (more) stage(sel ^ 1, k0 + 32);
#if HAVE_ASYNC_LDS
        if (more) ASYNC_WAIT(4);   // 4 newer asyncs in flight; current buf done
        else      ASYNC_WAIT(0);
#endif
        __syncthreads();

        // ---- fragments in contiguous k order (k = half*16 + s), same for A and B
        v16h afrag[4], bfrag[2];
        #pragma unroll
        for (int i = 0; i < 4; ++i) {
            int m = waveM * 64 + i * 16 + (lane & 15);
            #pragma unroll
            for (int p = 0; p < 8; ++p) {
                // A 16-bit 16x32: VGPR p -> K = half*16 + p*2 (+0,+1)
                int kb = (hsel << 4) + (p << 1);
                afrag[i][2 * p]     = As[sel][m * 32 + kb];
                afrag[i][2 * p + 1] = As[sel][m * 32 + kb + 1];
            }
        }
        #pragma unroll
        for (int j = 0; j < 2; ++j) {
            // B 16-bit 32x16: lane half selects K 0..15 / 16..31, contiguous
            int n = waveN * 32 + j * 16 + (lane & 15);
            bfrag[j] = *(const v16h*)(&Bs[sel][n * 32 + (hsel << 4)]);
        }

        #pragma unroll
        for (int i = 0; i < 4; ++i)
            #pragma unroll
            for (int j = 0; j < 2; ++j)
                acc[i][j] = __builtin_amdgcn_wmma_f32_16x16x32_f16(
                    false, afrag[i], false, bfrag[j],
                    (short)0, acc[i][j], false, false);
        __syncthreads();
        sel ^= 1;
    }

    // ---- store: C/D f32 16x16 layout: VGPR r -> M = r + half*8, lane%16 -> N
    #pragma unroll
    for (int i = 0; i < 4; ++i) {
        #pragma unroll
        for (int j = 0; j < 2; ++j) {
            int col = bn + waveN * 32 + j * 16 + (lane & 15);
            if (col < N) {
                #pragma unroll
                for (int r = 0; r < 8; ++r) {
                    int row = bm + waveM * 64 + i * 16 + (hsel << 3) + r;
                    float v = acc[i][j][r];
                    if (HALF_OUT)
                        ((_Float16*)Cv)[(size_t)row * N + col] = (_Float16)v;
                    else
                        ((float*)Cv)[(size_t)row * N + col] = v;
                }
            }
        }
    }
}

// ---------------------------------------------------------------- conv + SiLU
__global__ __launch_bounds__(256)
void conv_silu_kernel(const _Float16* __restrict__ xz,
                      const float* __restrict__ cw, const float* __restrict__ cb,
                      _Float16* __restrict__ xconv) {
    int idx = blockIdx.x * 256 + threadIdx.x;     // over NROWS*DINNER
    int d   = idx & (DINNER - 1);
    int row = idx >> 11;
    int t   = row & (TLEN - 1);
    float acc = cb[d];
    #pragma unroll
    for (int j = 0; j < 4; ++j) {
        int tt = t - 3 + j;
        if (tt >= 0)
            acc += cw[d * 4 + j] *
                   (float)xz[(size_t)(row - 3 + j) * (2 * DINNER) + d];
    }
    acc = acc / (1.f + expf(-acc));               // SiLU
    xconv[(size_t)row * DINNER + d] = (_Float16)acc;
}

// ---------------------------------------------------------------- alpha
__global__ __launch_bounds__(256)
void alpha_kernel(const float* __restrict__ bssm, const float* __restrict__ dtw,
                  const float* __restrict__ dtb, float* __restrict__ alpha) {
    __shared__ float red[256];
    __shared__ float bv[16];
    int row = blockIdx.x;
    int tid = threadIdx.x;
    if (tid < 16) bv[tid] = bssm[row * 16 + tid];
    __syncthreads();
    float local = 0.f;
    #pragma unroll
    for (int c = 0; c < 8; ++c) {
        int d = tid * 8 + c;
        float s = dtb[d];
        #pragma unroll
        for (int n = 0; n < 16; ++n) s += bv[n] * dtw[d * 16 + n];
        local += (s > 20.f) ? s : log1pf(expf(s));  // softplus
    }
    red[tid] = local;
    __syncthreads();
    for (int off = 128; off > 0; off >>= 1) {
        if (tid < off) red[tid] += red[tid + off];
        __syncthreads();
    }
    if (tid == 0) {
        float m = red[0] / (float)DINNER;
        alpha[row] = 1.f / (1.f + expf(-m));
    }
}

// ---------------------------------------------------------------- scan + gate
__global__ __launch_bounds__(256)
void scan_gate_kernel(const _Float16* __restrict__ xconv,
                      const _Float16* __restrict__ xz,
                      const float* __restrict__ alpha,
                      const float* __restrict__ Dp,
                      _Float16* __restrict__ yh) {
    int chan = blockIdx.x * 256 + threadIdx.x;    // 0..8191
    int b = chan >> 11;
    int d = chan & (DINNER - 1);
    float Dv = Dp[d];
    float h = 0.f;
    for (int t = 0; t < TLEN; ++t) {
        int row  = b * TLEN + t;
        float a  = alpha[row];
        float xv = (float)xconv[(size_t)row * DINNER + d];
        h = a * h + (1.f - a) * xv;
        float y  = h * Dv + xv;
        float zv = (float)xz[(size_t)row * (2 * DINNER) + DINNER + d];
        y *= zv / (1.f + expf(-zv));              // * silu(z)
        yh[(size_t)row * DINNER + d] = (_Float16)y;
    }
}

// ---------------------------------------------------------------- launch
extern "C" void kernel_launch(void* const* d_in, const int* in_sizes, int n_in,
                              void* d_out, int out_size, void* d_ws, size_t ws_size,
                              hipStream_t stream) {
    (void)in_sizes; (void)n_in; (void)out_size; (void)ws_size;
    const float* x     = (const float*)d_in[0];   // [8192,1024]
    const float* w_in  = (const float*)d_in[1];   // [4096,1024]
    const float* cw    = (const float*)d_in[2];   // [2048,1,4]
    const float* cb    = (const float*)d_in[3];   // [2048]
    const float* w_xp  = (const float*)d_in[4];   // [32,2048]
    const float* dtw   = (const float*)d_in[5];   // [2048,16]
    const float* dtb   = (const float*)d_in[6];   // [2048]
    const float* Dp    = (const float*)d_in[7];   // [2048]
    const float* w_out = (const float*)d_in[8];   // [1024,2048]
    float* out = (float*)d_out;                   // [8192,1024] f32

    char* w = (char*)d_ws;
    size_t off = 0;
    _Float16* x_h     = (_Float16*)(w + off); off += (size_t)NROWS * DMODEL * 2;       // 16 MB
    _Float16* w_in_h  = (_Float16*)(w + off); off += (size_t)4096 * 1024 * 2;          //  8 MB
    _Float16* w_out_h = (_Float16*)(w + off); off += (size_t)1024 * 2048 * 2;          //  4 MB
    _Float16* w_xp_h  = (_Float16*)(w + off); off += (size_t)16 * 2048 * 2;            // 64 KB
    _Float16* xz_h    = (_Float16*)(w + off); off += (size_t)NROWS * 4096 * 2;         // 64 MB
    _Float16* xconv_h = (_Float16*)(w + off); off += (size_t)NROWS * DINNER * 2;       // 32 MB
    float*    bssm    = (float*)(w + off);    off += (size_t)NROWS * 16 * 4;           // 0.5 MB
    float*    alpha   = (float*)(w + off);    off += (size_t)NROWS * 4;                // 32 KB
    _Float16* y_h     = (_Float16*)(w + off); off += (size_t)NROWS * DINNER * 2;       // 32 MB

    // f32 -> f16 conversions
    int n;
    n = NROWS * DMODEL;  cvt_f32_f16<<<(n + 255) / 256, 256, 0, stream>>>(x,     x_h,     n);
    n = 4096 * 1024;     cvt_f32_f16<<<(n + 255) / 256, 256, 0, stream>>>(w_in,  w_in_h,  n);
    n = 1024 * 2048;     cvt_f32_f16<<<(n + 255) / 256, 256, 0, stream>>>(w_out, w_out_h, n);
    n = 16 * 2048;       cvt_f32_f16<<<(n + 255) / 256, 256, 0, stream>>>(w_xp,  w_xp_h,  n);

    // in_proj: xz[8192,4096] = x_h @ w_in^T
    dim3 g1(4096 / 128, NROWS / 128);
    gemm_f16_wmma<true><<<g1, 256, 0, stream>>>(x_h, w_in_h, xz_h, NROWS, 4096, DMODEL);

    // depthwise causal conv + SiLU
    conv_silu_kernel<<<(NROWS * DINNER) / 256, 256, 0, stream>>>(xz_h, cw, cb, xconv_h);

    // B_ssm[8192,16] = x_conv @ x_proj_w[:16]^T   (C_ssm is dead code in reference)
    dim3 g2(1, NROWS / 128);
    gemm_f16_wmma<false><<<g2, 256, 0, stream>>>(xconv_h, w_xp_h, bssm, NROWS, 16, DINNER);

    // alpha[8192] = sigmoid(mean softplus(dt))
    alpha_kernel<<<NROWS, 256, 0, stream>>>(bssm, dtw, dtb, alpha);

    // sequential EMA scan + D-skip + SiLU(z) gate
    scan_gate_kernel<<<(BATCH * DINNER) / 256, 256, 0, stream>>>(xconv_h, xz_h, alpha, Dp, y_h);

    // out_proj: out[8192,1024] = y @ w_out^T  (f32 output)
    dim3 g3(1024 / 128, NROWS / 128);
    gemm_f16_wmma<false><<<g3, 256, 0, stream>>>(y_h, w_out_h, out, NROWS, 1024, DINNER);
}
